// Self_Attention_Block_52828097741287
// MI455X (gfx1250) — compile-verified, matrix operand order checked by reading, not run on hardware
//
#include <hip/hip_runtime.h>
#include <hip/hip_bf16.h>

// ---------------- CDNA5 WMMA types ----------------
typedef __attribute__((ext_vector_type(16))) __bf16       v16bf;
typedef __attribute__((ext_vector_type(8)))  float        v8f;
typedef __attribute__((ext_vector_type(4)))  unsigned int u32x4;

#define BATCH 4
#define DM    160           // d_model
#define NH    4
#define HD    40            // head dim
#define DP    64            // padded head dim (4 x 16)
#define HW    4096          // 64*64 tokens
#define SEQ   4097          // tokens + pad row
#define SEQ_Q 4096          // queries that survive cat[:, :-1, :]
#define SPAD  4128          // K/V padded length (multiple of 32)
#define FPAD  4112          // f rows padded   (multiple of 16)
#define SBLK  257           // ceil(SEQ/16)
#define KSTR  72            // LDS K-tile row stride (bank-conflict-free for b128)
#define VSTR  40            // LDS V/P-tile row stride (bank-conflict-free for b128)

union BFrag { v16bf v; u32x4 u[2]; };

// 16x32 bf16 A/B fragment loader (row-major tile, `stride` elements per row).
// Lanes 0-15 : row=lane,    K chunks {0..7} and {16..23}
// Lanes 16-31: row=lane-16, K chunks {8..15} and {24..31}
__device__ __forceinline__ BFrag ld_frag(const __bf16* base, int stride) {
  const int lane = threadIdx.x & 31;
  const int r  = lane & 15;
  const int hh = lane >> 4;
  const __bf16* p = base + (long)r * stride + 8 * hh;
  BFrag f;
  f.u[0] = *reinterpret_cast<const u32x4*>(p);
  f.u[1] = *reinterpret_cast<const u32x4*>(p + 16);
  return f;
}

__device__ __forceinline__ v8f wmma_bf16(const BFrag& a, const BFrag& b, v8f c) {
  return __builtin_amdgcn_wmma_f32_16x16x32_bf16(false, a.v, false, b.v,
                                                 (short)0, c, false, false);
}

__device__ __forceinline__ v8f vzero() {
  v8f z = {0.f, 0.f, 0.f, 0.f, 0.f, 0.f, 0.f, 0.f};
  return z;
}

// ---------------- scratch zero (q/k/vt pad rows) ----------------
__global__ void k_zero(float4* __restrict__ p, long n) {
  long i = (long)blockIdx.x * blockDim.x + threadIdx.x;
  if (i < n) p[i] = make_float4(0.f, 0.f, 0.f, 0.f);
}

// ---------------- ones row in Vt at d==HD (softmax denominator via WMMA) ----------------
__global__ void k_ones(__bf16* __restrict__ vt) {
  int i = blockIdx.x * 256 + threadIdx.x;
  if (i >= 16 * SPAD) return;
  int bh = i / SPAD, s = i % SPAD;
  vt[((long)bh * DP + HD) * SPAD + s] = (__bf16)1.0f;
}

// ---------------- conv1: 64->64, 2x2 stride 2 ----------------
__global__ void k_conv1(const float* __restrict__ x, const float* __restrict__ w,
                        const float* __restrict__ bias, float* __restrict__ c1) {
  int idx = blockIdx.x * 256 + threadIdx.x;
  if (idx >= BATCH * 64 * HW) return;
  int ow = idx & 63;
  int oh = (idx >> 6) & 63;
  int oc = (idx >> 12) & 63;
  int b  = idx >> 18;
  const float* xb = x + (long)b * 64 * 128 * 128;
  const float* wr = w + oc * 64 * 4;
  float s = bias[oc];
  #pragma unroll 4
  for (int ic = 0; ic < 64; ++ic) {
    const float* xp = xb + ((long)ic * 128 + 2 * oh) * 128 + 2 * ow;
    const float* wp = wr + ic * 4;
    s += xp[0] * wp[0] + xp[1] * wp[1] + xp[128] * wp[2] + xp[129] * wp[3];
  }
  c1[((long)b * 64 + oc) * HW + (oh * 64 + ow)] = s;
}

// ---------- conv2 (1x1) + bias + leaky + transpose + pad + PE -> bf16 tokens ----------
__global__ void k_conv2f(const float* __restrict__ c1, const float* __restrict__ w,
                         const float* __restrict__ bias, const float* __restrict__ pe,
                         __bf16* __restrict__ fbf) {
  int idx = blockIdx.x * 256 + threadIdx.x;
  if (idx >= BATCH * FPAD * DM) return;
  int oc = idx % DM;
  int s  = (idx / DM) % FPAD;
  int b  = idx / (DM * FPAD);
  float v = 0.f;
  if (s < HW) {
    float acc = bias[oc];
    const float* cp = c1 + (long)b * 64 * HW + s;
    const float* wp = w + oc * 64;
    #pragma unroll 8
    for (int ic = 0; ic < 64; ++ic) acc += cp[(long)ic * HW] * wp[ic];
    acc = acc >= 0.f ? acc : 0.01f * acc;
    v = acc + pe[s];
  } else if (s == HW) {
    v = pe[HW];            // zero-pad row + positional value
  }                        // rows 4097..4111 stay 0 (safe WMMA tiles)
  fbf[((long)b * FPAD + s) * DM + oc] = (__bf16)v;
}

// ---------------- wq/wk/wv -> bf16 ----------------
__global__ void k_cvtw(const float* __restrict__ wq, const float* __restrict__ wk,
                       const float* __restrict__ wv, __bf16* __restrict__ o) {
  int i = blockIdx.x * 256 + threadIdx.x;
  if (i >= 3 * DM * DM) return;
  const float* src = (i < DM * DM) ? wq : (i < 2 * DM * DM ? wk : wv);
  o[i] = (__bf16)src[i % (DM * DM)];
}

// ---------------- QKV projection GEMM (WMMA bf16) ----------------
// one wave = one 16x16 tile of one of {q,k,v}; K = 160 = 5 x K32 steps
// Q output is pre-scaled by (1/sqrt(HD)) * log2(e) so attention uses raw exp2.
__global__ void __launch_bounds__(256) k_qkv(const __bf16* __restrict__ fbf,
                                             const __bf16* __restrict__ wb,
                                             __bf16* __restrict__ q,
                                             __bf16* __restrict__ k,
                                             __bf16* __restrict__ vt) {
  const int wv_id = blockIdx.x * 8 + (threadIdx.x >> 5);
  if (wv_id >= 3 * BATCH * SBLK * 10) return;
  int t = wv_id;
  const int mat = t % 3;  t /= 3;
  const int ob  = t % 10; t /= 10;
  const int sb  = t % SBLK;
  const int b   = t / SBLK;

  const __bf16* A0 = fbf + ((long)b * FPAD + sb * 16) * DM;
  const __bf16* B0 = wb + (long)mat * DM * DM + (long)ob * 16 * DM;

  v8f acc = vzero();
  #pragma unroll
  for (int kk = 0; kk < 5; ++kk) {
    BFrag a  = ld_frag(A0 + kk * 32, DM);
    BFrag bb = ld_frag(B0 + kk * 32, DM);
    acc = wmma_bf16(a, bb, acc);
  }

  const int lane = threadIdx.x & 31;
  const int n  = lane & 15;
  const int hm = lane >> 4;
  const int o  = ob * 16 + n;
  const int hh = o / HD;
  const int d  = o - hh * HD;
  const long bh = (long)b * NH + hh;
  const float qscale = 0.22811012f;   // (1/sqrt(40)) * log2(e)
  #pragma unroll
  for (int i = 0; i < 8; ++i) {
    int s = sb * 16 + i + 8 * hm;     // < 4112 < SPAD
    float av = acc[i];
    if (mat == 0) av *= qscale;
    __bf16 val = (__bf16)av;
    if (mat == 0)      q [(bh * SPAD + s) * DP + d] = val;
    else if (mat == 1) k [(bh * SPAD + s) * DP + d] = val;
    else               vt[(bh * DP + d) * SPAD + s] = val;   // V stored transposed
  }
}

// ---------------- flash attention (WMMA bf16, online softmax) ----------------
// 1 block = 8 waves = 8 consecutive 16-query tiles of one (b,h).
// K/V tiles cooperatively double-buffered in LDS; softmax denominator comes out
// of the P*Vt WMMA via the ones-row at d==HD.
__global__ void __launch_bounds__(256) k_attn(const __bf16* __restrict__ q,
                                              const __bf16* __restrict__ k,
                                              const __bf16* __restrict__ vt,
                                              float* __restrict__ cat) {
  __shared__ __attribute__((aligned(16))) __bf16 sK[2][32 * KSTR];
  __shared__ __attribute__((aligned(16))) __bf16 sV[2][64 * VSTR];
  __shared__ __attribute__((aligned(16))) __bf16 sP[8][16 * VSTR];

  const int bh = blockIdx.x >> 5;              // 16 (b,h) pairs
  const int qg = blockIdx.x & 31;              // 32 groups of 128 queries
  const int wv = threadIdx.x >> 5;
  const int qb = qg * 8 + wv;                  // 16-query tile index (< 256)
  const int lane = threadIdx.x & 31;
  const int n  = lane & 15;
  const int hm = lane >> 4;
  const int tid = threadIdx.x;
  __bf16* pb = &sP[wv][0];

  // cooperative staging coordinates (256 threads: K 32x64, V 64x32, 16B each)
  const int krow = tid >> 3, kc8 = (tid & 7) * 8;
  const int vrow = tid >> 2, vc8 = (tid & 3) * 8;
  const __bf16* kg = k  + ((long)bh * SPAD + krow) * DP + kc8;
  const __bf16* vg = vt + ((long)bh * DP + vrow) * SPAD + vc8;

  const __bf16* qbase = q + ((long)bh * SPAD + qb * 16) * DP;
  BFrag Qa = ld_frag(qbase, DP);        // K = d 0..31
  BFrag Qb = ld_frag(qbase + 32, DP);   // K = d 32..63

  v8f acc0 = vzero(), acc1 = vzero(), acc2 = vzero(), acc3 = vzero();
  float mrow[8];
  #pragma unroll
  for (int r = 0; r < 8; ++r) mrow[r] = -1e30f;

  // stage tile 0
  {
    u32x4 a = *reinterpret_cast<const u32x4*>(kg);
    u32x4 b = *reinterpret_cast<const u32x4*>(vg);
    *reinterpret_cast<u32x4*>(&sK[0][krow * KSTR + kc8]) = a;
    *reinterpret_cast<u32x4*>(&sV[0][vrow * VSTR + vc8]) = b;
  }
  __syncthreads();

  const int NIT = (SEQ + 31) / 32;             // 129
  for (int it = 0; it < NIT; ++it) {
    const int j0 = it * 32;
    const int cur = it & 1;
    const bool more = (it + 1 < NIT);
    u32x4 ka, va;
    if (more) {                                // prefetch next tile into regs
      ka = *reinterpret_cast<const u32x4*>(kg + (long)(j0 + 32) * DP);
      va = *reinterpret_cast<const u32x4*>(vg + (j0 + 32));
      if (it + 2 < NIT) {
        __builtin_prefetch(kg + (long)(j0 + 64) * DP, 0, 0);
        __builtin_prefetch(vg + (j0 + 64), 0, 0);
      }
    }

    // scores: Q (16x64) x K-tile (32x64)^T  -> two 16x16 f32 fragments
    const __bf16* kt = &sK[cur][0];
    BFrag K1a = ld_frag(kt, KSTR);
    BFrag K1b = ld_frag(kt + 32, KSTR);
    BFrag K2a = ld_frag(kt + 16 * KSTR, KSTR);
    BFrag K2b = ld_frag(kt + 16 * KSTR + 32, KSTR);
    v8f S1 = wmma_bf16(Qb, K1b, wmma_bf16(Qa, K1a, vzero()));
    v8f S2 = wmma_bf16(Qb, K2b, wmma_bf16(Qa, K2a, vzero()));

    const bool msk1 = (j0 + n)      >= SEQ;    // one column per lane
    const bool msk2 = (j0 + 16 + n) >= SEQ;

    float p1[8], p2[8];
    #pragma unroll
    for (int r = 0; r < 8; ++r) {
      float a1 = msk1 ? -1e30f : S1[r];        // already in exp2 domain
      float a2 = msk2 ? -1e30f : S2[r];
      float mx = fmaxf(a1, a2);                // row max over 16-lane half
      mx = fmaxf(mx, __shfl_xor(mx, 1, 32));
      mx = fmaxf(mx, __shfl_xor(mx, 2, 32));
      mx = fmaxf(mx, __shfl_xor(mx, 4, 32));
      mx = fmaxf(mx, __shfl_xor(mx, 8, 32));
      float mn   = fmaxf(mrow[r], mx);
      float corr = __builtin_amdgcn_exp2f(mrow[r] - mn);
      mrow[r] = mn;
      p1[r] = __builtin_amdgcn_exp2f(a1 - mn);
      p2[r] = __builtin_amdgcn_exp2f(a2 - mn);
      acc0[r] *= corr; acc1[r] *= corr; acc2[r] *= corr; acc3[r] *= corr;
    }

    // C-layout f32 P -> bf16 A-layout via per-wave LDS staging tile
    #pragma unroll
    for (int r = 0; r < 8; ++r) {
      int m = r + 8 * hm;
      pb[m * VSTR + n]      = (__bf16)p1[r];
      pb[m * VSTR + 16 + n] = (__bf16)p2[r];
    }
    asm volatile("s_wait_dscnt 0" ::: "memory");
    BFrag P = ld_frag(pb, VSTR);

    // P (16x32) x Vt-tile rows (64 x 32); row d==40 of Vt is all-ones -> l in acc2
    const __bf16* vtile = &sV[cur][0];
    acc0 = wmma_bf16(P, ld_frag(vtile,                VSTR), acc0);
    acc1 = wmma_bf16(P, ld_frag(vtile + 16 * VSTR,    VSTR), acc1);
    acc2 = wmma_bf16(P, ld_frag(vtile + 32 * VSTR,    VSTR), acc2);
    acc3 = wmma_bf16(P, ld_frag(vtile + 48 * VSTR,    VSTR), acc3);

    if (more) {                                // fill the other buffer
      *reinterpret_cast<u32x4*>(&sK[1 - cur][krow * KSTR + kc8]) = ka;
      *reinterpret_cast<u32x4*>(&sV[1 - cur][vrow * VSTR + vc8]) = va;
    }
    __syncthreads();
  }

  // softmax denominator = column d==HD (fragment 2, n==8); broadcast per half
  float linv[8];
  #pragma unroll
  for (int r = 0; r < 8; ++r) {
    float l = __shfl(acc2[r], (lane & 16) | 8, 32);
    linv[r] = __builtin_amdgcn_rcpf(l);
  }

  const int  hh   = bh & (NH - 1);
  const long brow = (long)(bh >> 2) * SEQ_Q;
  v8f accar[4] = {acc0, acc1, acc2, acc3};
  #pragma unroll
  for (int f = 0; f < 4; ++f) {
    int dl = f * 16 + n;
    if (dl < HD) {
      v8f a = accar[f];
      #pragma unroll
      for (int r = 0; r < 8; ++r) {
        int m = qb * 16 + r + 8 * hm;               // < 4096
        cat[(brow + m) * DM + hh * HD + dl] = a[r] * linv[r];
      }
    }
  }
}

// ---------------- LayerNorm + residual + transpose + 2x nearest upsample ----------------
__global__ void k_epi(const float* __restrict__ cat, const float* __restrict__ lw,
                      const float* __restrict__ lb, float* __restrict__ out) {
  int idx = blockIdx.x * 256 + threadIdx.x;
  if (idx >= BATCH * HW) return;
  int s = idx % HW, b = idx / HW;
  const float* row = cat + ((long)b * HW + s) * DM;
  float mu = 0.f;
  for (int c = 0; c < DM; ++c) mu += row[c];
  mu *= (1.f / DM);
  float var = 0.f;
  for (int c = 0; c < DM; ++c) { float d = row[c] - mu; var += d * d; }
  var *= (1.f / DM);
  float inv = rsqrtf(var + 1e-5f);
  int ih = s >> 6, iw = s & 63;
  for (int c = 0; c < DM; ++c) {
    float xv = row[c];
    float v = (xv - mu) * inv * lw[c] + lb[c] + xv;
    float* o = out + (((long)b * DM + c) * 128 + 2 * ih) * 128 + 2 * iw;
    o[0] = v; o[1] = v; o[128] = v; o[129] = v;
  }
}

// ---------------- launch ----------------
extern "C" void kernel_launch(void* const* d_in, const int* in_sizes, int n_in,
                              void* d_out, int out_size, void* d_ws, size_t ws_size,
                              hipStream_t stream) {
  (void)in_sizes; (void)n_in; (void)out_size; (void)ws_size;
  const float* x   = (const float*)d_in[0];
  const float* w1  = (const float*)d_in[1];
  const float* b1  = (const float*)d_in[2];
  const float* w2  = (const float*)d_in[3];
  const float* b2  = (const float*)d_in[4];
  const float* pe  = (const float*)d_in[5];
  const float* wq  = (const float*)d_in[6];
  const float* wk  = (const float*)d_in[7];
  const float* wv  = (const float*)d_in[8];
  const float* lw  = (const float*)d_in[9];
  const float* lb  = (const float*)d_in[10];
  float* out = (float*)d_out;
  char*  ws  = (char*)d_ws;

  // workspace layout (bytes), all 16B aligned
  constexpr long C1_OFF  = 0;                                   // 4*64*4096 f32      = 4,194,304 B
  constexpr long FBF_OFF = C1_OFF  + 4L * 64 * HW * 4;          // 4*4112*160 bf16    = 5,263,360 B
  constexpr long WB_OFF  = FBF_OFF + (long)BATCH * FPAD * DM * 2;   // 3*160*160 bf16 = 153,600 B
  constexpr long Q_OFF   = WB_OFF  + 3L * DM * DM * 2;          // 16*4128*64 bf16    = 8,454,144 B
  constexpr long K_OFF   = Q_OFF   + 16L * SPAD * DP * 2;
  constexpr long VT_OFF  = K_OFF   + 16L * SPAD * DP * 2;
  constexpr long CAT_OFF = VT_OFF  + 16L * SPAD * DP * 2;       // 4*4096*160 f32     = 10,485,760 B

  float*  c1  = (float*) (ws + C1_OFF);
  __bf16* fbf = (__bf16*)(ws + FBF_OFF);
  __bf16* wb  = (__bf16*)(ws + WB_OFF);
  __bf16* qb  = (__bf16*)(ws + Q_OFF);
  __bf16* kb  = (__bf16*)(ws + K_OFF);
  __bf16* vtb = (__bf16*)(ws + VT_OFF);
  float*  cat = (float*) (ws + CAT_OFF);

  // zero q/k/vt (contiguous) so pad rows are well-defined every call
  const long zero16 = (3L * 16 * SPAD * DP * 2) / 16;           // float4 count
  k_zero <<<(int)((zero16 + 255) / 256), 256, 0, stream>>>((float4*)qb, zero16);
  k_ones <<<(16 * SPAD + 255) / 256, 256, 0, stream>>>(vtb);

  k_cvtw <<<(3 * DM * DM + 255) / 256, 256, 0, stream>>>(wq, wk, wv, wb);
  k_conv1<<<(BATCH * 64 * HW) / 256, 256, 0, stream>>>(x, w1, b1, c1);
  k_conv2f<<<(BATCH * FPAD * DM + 255) / 256, 256, 0, stream>>>(c1, w2, b2, pe, fbf);

  const int qkv_waves = 3 * BATCH * SBLK * 10;                  // 30840
  k_qkv  <<<(qkv_waves + 7) / 8, 256, 0, stream>>>(fbf, wb, qb, kb, vtb);

  k_attn <<<512, 256, 0, stream>>>(qb, kb, vtb, cat);           // 16 bh x 32 q-groups

  k_epi  <<<(BATCH * HW) / 256, 256, 0, stream>>>(cat, lw, lb, out);
}
